// ClipQuantized_22832046145536
// MI455X (gfx1250) — matrix operand order repaired and checked by reading, not compiled, not run on hardware
//
#include <hip/hip_runtime.h>

// ---- Problem constants (from reference) ----
#define NVEC 8192     // rows per side
#define DDIM 512      // embedding dim
#define MSUB 32       // subspaces
#define KCEN 256      // centroids per subspace
#define DSUB 16       // dim per subspace
#define TDIM 8192     // text rows (== NVEC)

typedef __attribute__((ext_vector_type(16))) _Float16 v16h;
typedef __attribute__((ext_vector_type(8)))  _Float16 v8h;
typedef __attribute__((ext_vector_type(8)))  float    v8f;

union HV { v16h v; v8h h[2]; };

// ============================================================
// Kernel 1: PQ encode (exact f32), emit f16 quantized vectors +
// deterministic per-block loss partials.
// grid = (NVEC/256, MSUB, 2), block = 256
// ============================================================
__global__ void __launch_bounds__(256)
pq_encode(const float* __restrict__ img, const float* __restrict__ txt,
          const float* __restrict__ centroids,
          _Float16* __restrict__ img_q, _Float16* __restrict__ txt_q,
          float* __restrict__ partials)
{
    __shared__ float Cs[KCEN][DSUB];   // 16 KB
    __shared__ float cn[KCEN];
    __shared__ float red[256];

    const int t    = threadIdx.x;
    const int m    = blockIdx.y;
    const int side = blockIdx.z;

    const float*  vec = side ? txt : img;
    _Float16*     q   = side ? txt_q : img_q;

    // load centroid row k = t (16 floats) into LDS, compute ||c||^2
    {
        const float4* cp = (const float4*)(centroids + ((size_t)m * KCEN + t) * DSUB);
        float4 c0 = cp[0], c1 = cp[1], c2 = cp[2], c3 = cp[3];
        float4* dst = (float4*)Cs[t];
        dst[0] = c0; dst[1] = c1; dst[2] = c2; dst[3] = c3;
        float s = c0.x*c0.x + c0.y*c0.y + c0.z*c0.z + c0.w*c0.w
                + c1.x*c1.x + c1.y*c1.y + c1.z*c1.z + c1.w*c1.w
                + c2.x*c2.x + c2.y*c2.y + c2.z*c2.z + c2.w*c2.w
                + c3.x*c3.x + c3.y*c3.y + c3.z*c3.z + c3.w*c3.w;
        cn[t] = s;
    }
    __syncthreads();

    const int n = blockIdx.x * 256 + t;

    // load this thread's sub-vector (16 f32, contiguous)
    float v[DSUB];
    {
        const float4* vp = (const float4*)(vec + (size_t)n * DDIM + m * DSUB);
        float4 a = vp[0], b = vp[1], c = vp[2], d = vp[3];
        v[0]=a.x; v[1]=a.y; v[2]=a.z; v[3]=a.w;
        v[4]=b.x; v[5]=b.y; v[6]=b.z; v[7]=b.w;
        v[8]=c.x; v[9]=c.y; v[10]=c.z; v[11]=c.w;
        v[12]=d.x; v[13]=d.y; v[14]=d.z; v[15]=d.w;
    }
    float vnorm = 0.f;
    #pragma unroll
    for (int d = 0; d < DSUB; ++d) vnorm = fmaf(v[d], v[d], vnorm);

    // argmin_k ( ||c_k||^2 - 2 v.c_k )   (== argmin of full distance)
    float best = 3.4e38f;
    int   bk   = 0;
    for (int k = 0; k < KCEN; ++k) {
        float dot = 0.f;
        #pragma unroll
        for (int d = 0; d < DSUB; ++d) dot = fmaf(v[d], Cs[k][d], dot);
        float sc = cn[k] - 2.0f * dot;
        if (sc < best) { best = sc; bk = k; }   // strict '<' keeps first index like jnp.argmin
    }

    // write quantized sub-vector in f16
    {
        _Float16* qp = q + (size_t)n * DDIM + m * DSUB;
        #pragma unroll
        for (int d = 0; d < DSUB; ++d) qp[d] = (_Float16)Cs[bk][d];
    }

    // loss partial: ||q - v||^2 = best + ||v||^2 ; loss = (ALPHA+BETA)/N * sum
    red[t] = best + vnorm;
    __syncthreads();
    for (int off = 128; off > 0; off >>= 1) {
        if (t < off) red[t] += red[t + off];
        __syncthreads();
    }
    if (t == 0) {
        int bid = blockIdx.x + gridDim.x * (blockIdx.y + gridDim.y * blockIdx.z);
        partials[bid] = (2.0f / (float)NVEC) * red[0];
    }
}

// ============================================================
// Kernel 2: deterministic fixed-order loss reduction (2048 partials)
// ============================================================
__global__ void __launch_bounds__(256)
loss_reduce(const float* __restrict__ partials, float* __restrict__ out_loss)
{
    __shared__ float red[256];
    const int t = threadIdx.x;
    float s = 0.f;
    for (int i = t; i < 2048; i += 256) s += partials[i];
    red[t] = s;
    __syncthreads();
    for (int off = 128; off > 0; off >>= 1) {
        if (t < off) red[t] += red[t + off];
        __syncthreads();
    }
    if (t == 0) *out_loss = red[0];
}

// ============================================================
// Kernel 3: logits = 100 * (img_q @ txt_q^T) via v_wmma_f32_16x16x32_f16
// block = 256 (8 waves), block tile 128x256, wave tile 64x64 (4x4 WMMA)
// grid = (TDIM/256, NVEC/128)
// A-frag: lane holds M=lane&15, two 8-half chunks
//   at K = (lane>>4)*16 and (lane>>4)*16 + 8 (contiguous, matches B).
// B-frag (K x N): lane holds N=lane&15, one contiguous 16-half chunk at
//   K = (lane>>4)*16  -> a row of txt_q, no transpose needed.
// ============================================================
__global__ void __launch_bounds__(256)
pq_logits_wmma(const _Float16* __restrict__ A, const _Float16* __restrict__ B,
               float* __restrict__ out)
{
    const int t  = threadIdx.x;
    const int L  = t & 31;
    const int w  = t >> 5;
    const int lm = L & 15;
    const int lh = L >> 4;
    const int wr = w >> 2;        // 0..1
    const int wc = w & 3;         // 0..3

    const int rowBase = blockIdx.y * 128 + wr * 64;
    const int colBase = blockIdx.x * 256 + wc * 64;

    v8f acc[4][4] = {};

    const _Float16* aBase[4];
    const _Float16* bBase[4];
    #pragma unroll
    for (int i = 0; i < 4; ++i)
        aBase[i] = A + (size_t)(rowBase + i * 16 + lm) * DDIM + lh * 16;
    #pragma unroll
    for (int j = 0; j < 4; ++j)
        bBase[j] = B + (size_t)(colBase + j * 16 + lm) * DDIM + lh * 16;

    for (int ks = 0; ks < DDIM / 32; ++ks) {
        const int k0 = ks * 32;
        HV a[4], b[4];
        #pragma unroll
        for (int i = 0; i < 4; ++i) {
            a[i].h[0] = *(const v8h*)(aBase[i] + k0);
            a[i].h[1] = *(const v8h*)(aBase[i] + k0 + 8);
        }
        #pragma unroll
        for (int j = 0; j < 4; ++j) {
            b[j].v = *(const v16h*)(bBase[j] + k0);
        }
        #pragma unroll
        for (int i = 0; i < 4; ++i) {
            #pragma unroll
            for (int j = 0; j < 4; ++j) {
                acc[i][j] = __builtin_amdgcn_wmma_f32_16x16x32_f16(
                    false, a[i].v, false, b[j].v,
                    (short)0, acc[i][j], false, false);
            }
        }
    }

    // C/D layout: VGPR r, lane L -> row = r + 8*(L>>4), col = L&15
    #pragma unroll
    for (int i = 0; i < 4; ++i) {
        #pragma unroll
        for (int j = 0; j < 4; ++j) {
            const int row = rowBase + i * 16 + lh * 8;
            const int col = colBase + j * 16 + lm;
            float* op = out + (size_t)row * TDIM + col;
            #pragma unroll
            for (int r = 0; r < 8; ++r)
                op[(size_t)r * TDIM] = 100.0f * acc[i][j][r];
        }
    }
}

// ============================================================
// Kernel 4: per-row max then sum(exp(x-max)).  grid = NVEC rows.
// ============================================================
__global__ void __launch_bounds__(256)
row_stats(const float* __restrict__ logits,
          float* __restrict__ rmax, float* __restrict__ rsum)
{
    __shared__ float sm[256];
    const int row = blockIdx.x;
    const int t   = threadIdx.x;
    const float* rp = logits + (size_t)row * TDIM;

    float m = -3.4e38f;
    for (int j = t; j < TDIM; j += 256) m = fmaxf(m, rp[j]);
    sm[t] = m;
    __syncthreads();
    for (int off = 128; off > 0; off >>= 1) {
        if (t < off) sm[t] = fmaxf(sm[t], sm[t + off]);
        __syncthreads();
    }
    const float rowm = sm[0];
    __syncthreads();

    float s = 0.f;
    for (int j = t; j < TDIM; j += 256) s += __expf(rp[j] - rowm);
    sm[t] = s;
    __syncthreads();
    for (int off = 128; off > 0; off >>= 1) {
        if (t < off) sm[t] += sm[t + off];
        __syncthreads();
    }
    if (t == 0) { rmax[row] = rowm; rsum[row] = sm[0]; }
}

// ============================================================
// Kernel 5: in-place softmax normalize, float4 vectorized.
// grid = NVEC*TDIM/4/256 = 65536
// ============================================================
__global__ void __launch_bounds__(256)
softmax_norm(float* __restrict__ out,
             const float* __restrict__ rmax, const float* __restrict__ rsum)
{
    const size_t i4   = (size_t)blockIdx.x * 256 + threadIdx.x;
    const size_t base = i4 * 4;
    const int row = (int)(base >> 13);          // / 8192
    const float mx  = rmax[row];
    const float inv = 1.0f / rsum[row];
    float4 x = *(float4*)(out + base);
    x.x = __expf(x.x - mx) * inv;
    x.y = __expf(x.y - mx) * inv;
    x.z = __expf(x.z - mx) * inv;
    x.w = __expf(x.w - mx) * inv;
    *(float4*)(out + base) = x;
}

// ============================================================
// Host-side launcher
// ============================================================
extern "C" void kernel_launch(void* const* d_in, const int* in_sizes, int n_in,
                              void* d_out, int out_size, void* d_ws, size_t ws_size,
                              hipStream_t stream)
{
    const float* img = (const float*)d_in[0];   // [8192, 512]
    const float* txt = (const float*)d_in[1];   // [8192, 512]
    const float* cen = (const float*)d_in[2];   // [32, 256, 16]

    float* out = (float*)d_out;                 // [8192*8192] logits/softmax + [1] loss

    // workspace layout
    char* ws = (char*)d_ws;
    _Float16* img_q = (_Float16*)(ws);                                   // 8 MB
    _Float16* txt_q = (_Float16*)(ws + (size_t)8 * 1024 * 1024);         // 8 MB
    float*    rmx   = (float*)(ws + (size_t)16 * 1024 * 1024);           // 32 KB
    float*    rsm   = (float*)(ws + (size_t)16 * 1024 * 1024 + 32768);   // 32 KB
    float*    part  = (float*)(ws + (size_t)16 * 1024 * 1024 + 65536);   // 8 KB

    float* loss = out + (size_t)NVEC * TDIM;

    // 1) PQ encode both sides, produce f16 quantized vectors + loss partials
    dim3 egrid(NVEC / 256, MSUB, 2);
    pq_encode<<<egrid, 256, 0, stream>>>(img, txt, cen, img_q, txt_q, part);

    // 2) deterministic loss reduction
    loss_reduce<<<1, 256, 0, stream>>>(part, loss);

    // 3) 8192x8192x512 GEMM in f16 WMMA -> scaled logits into d_out
    dim3 ggrid(TDIM / 256, NVEC / 128);
    pq_logits_wmma<<<ggrid, 256, 0, stream>>>(img_q, txt_q, out);

    // 4) row-wise softmax stats
    row_stats<<<NVEC, 256, 0, stream>>>(out, rmx, rsm);

    // 5) normalize in place
    softmax_norm<<<(NVEC * (TDIM / 4)) / 256, 256, 0, stream>>>(out, rmx, rsm);
}
